// TransformerEncoderLayer_31516470018515
// MI455X (gfx1250) — matrix operand order repaired and checked by reading, not compiled, not run on hardware
//
#include <hip/hip_runtime.h>
#include <math.h>

// ---------------------------------------------------------------------------
// TransformerEncoderLayer for gfx1250 (MI455X, wave32, WMMA).
// GEMMs: v_wmma_f32_16x16x32_bf16, 32x32 per wave (2x2 fragments), 128x64
// block tiles, double-buffered LDS fed by TDM tensor_load_to_lds (TENSORcnt)
// or per-lane async global->LDS copies (ASYNCcnt) as fallback.
// ---------------------------------------------------------------------------

typedef __bf16 bf16;
typedef __attribute__((ext_vector_type(16))) __bf16 v16bf;
typedef __attribute__((ext_vector_type(8)))  float  v8f;
typedef __attribute__((ext_vector_type(8)))  __bf16 bf16x8;
typedef __attribute__((ext_vector_type(4)))  __bf16 bf16x4;
typedef __attribute__((ext_vector_type(4)))  int    v4i;
typedef __attribute__((ext_vector_type(8)))  int    v8i;
typedef __attribute__((ext_vector_type(4)))  unsigned int v4u;

constexpr int cB = 2, cS = 4096, cD = 512, cH = 8, cFF = 2048, cHD = 64;
constexpr int ROWS = cB * cS;                 // 8192 token rows
constexpr float cEPS = 1e-5f;

// ----------------------- async global->LDS copy ----------------------------
#if defined(__gfx1250__) && __has_builtin(__builtin_amdgcn_global_load_async_to_lds_b128)
#define USE_ASYNC_COPY 1
#else
#define USE_ASYNC_COPY 0
#endif

#if defined(__gfx1250__) && __has_builtin(__builtin_amdgcn_tensor_load_to_lds)
#define USE_TDM 1
#else
#define USE_TDM 0
#endif

// Copy 16 bytes global -> LDS (per-lane). ASYNCcnt-tracked DMA when available.
__device__ __forceinline__ void copy16_g2s(const bf16* g, bf16* s) {
#if USE_ASYNC_COPY
  __builtin_amdgcn_global_load_async_to_lds_b128(
      (__attribute__((address_space(1))) v4i*)(bf16*)g,
      (__attribute__((address_space(3))) v4i*)s, 0, 0);
#else
  *(bf16x8*)s = *(const bf16x8*)g;
#endif
}

__device__ __forceinline__ void wait_async_copies() {
#if USE_ASYNC_COPY
#if __has_builtin(__builtin_amdgcn_s_wait_asynccnt)
  __builtin_amdgcn_s_wait_asynccnt(0);
#else
  asm volatile("s_wait_asynccnt 0x0" ::: "memory");
#endif
#endif
}

#if USE_TDM
__device__ __forceinline__ unsigned lds_byte_addr(const void* p) {
  return (unsigned)(size_t)(__attribute__((address_space(3))) const void*)p;
}

// Issue one TDM 2D tile load (bf16 elements), one instruction per wave.
// Tile: tile_d0 elems/row x tile_d1 rows; global row stride = stride0 elems.
// LDS padding: pad_amount DWORDs inserted every (1<<pad_interval) 8-byte units
// so padded LDS rows match our (+8 bf16) leading dimensions.
__device__ __forceinline__ void tdm_load_2d(const bf16* gsrc, unsigned lds,
                                            unsigned tile_d0, unsigned tile_d1,
                                            unsigned stride0,
                                            unsigned pad_interval,
                                            unsigned pad_amount) {
  const unsigned long long ga = (unsigned long long)(size_t)gsrc;
  v4u g0;
  g0[0] = 1u;                                       // count=1, user descriptor
  g0[1] = lds;                                      // lds_addr (bytes)
  g0[2] = (unsigned)(ga & 0xffffffffu);             // global_addr[31:0]
  g0[3] = (unsigned)((ga >> 32) & 0x01ffffffu) |    // global_addr[56:32]
          (2u << 30);                               // type = 2 (image)
  v8i g1;
  g1[0] = (int)((1u << 16) |                        // data_size = 1 (2 bytes)
                (1u << 20) |                        // pad_enable
                (pad_interval << 22) | (pad_amount << 25));
  g1[1] = (int)((tile_d0 & 0xffffu) << 16);         // tensor_dim0[15:0]
  g1[2] = (int)(((tile_d0 >> 16) & 0xffffu) |       // tensor_dim0[31:16]
                ((tile_d1 & 0xffffu) << 16));       // tensor_dim1[15:0]
  g1[3] = (int)(((tile_d1 >> 16) & 0xffffu) |       // tensor_dim1[31:16]
                ((tile_d0 & 0xffffu) << 16));       // tile_dim0
  g1[4] = (int)(tile_d1 & 0xffffu);                 // tile_dim1 (tile_dim2=0)
  g1[5] = (int)stride0;                             // tensor_dim0_stride[31:0]
  g1[6] = 0;                                        // stride0 hi / stride1 lo
  g1[7] = 0;                                        // stride1 hi
  const v4i gz4 = {0, 0, 0, 0};
  const v8i gz8 = {0, 0, 0, 0, 0, 0, 0, 0};
  __builtin_amdgcn_tensor_load_to_lds(g0, g1, gz4, gz4, gz8, 0);
}
#endif

__device__ __forceinline__ void wait_tiles() {
#if USE_TDM
  __builtin_amdgcn_s_wait_tensorcnt(0);
#else
  wait_async_copies();
#endif
}

// ----------------------------- WMMA helpers --------------------------------
// A fragment: 16x32 bf16, row-major source with leading dim lda.
// lanes 0-15 -> M=0..15 (K=0..15),
// lanes 16-31 -> M=0..15 (K=16..31).
__device__ __forceinline__ v16bf load_a_frag(const bf16* A, int lda) {
  const int lane = threadIdx.x & 31;
  const int m    = lane & 15;
  const int khi  = (lane >> 4) << 4;          // 0 or 16
  const bf16* row = A + m * lda;
  v16bf a;
#pragma unroll
  for (int j = 0; j < 8; ++j) {
    const int kb = 2 * j + khi;
    a[2 * j]     = row[kb];
    a[2 * j + 1] = row[kb + 1];
  }
  return a;
}

// B fragment: 32x16 bf16, row-major KxN source with leading dim ldb.
// lanes 0-15 hold K=0..15, lanes 16-31 hold K=16..31; N = lane & 15.
__device__ __forceinline__ v16bf load_b_frag(const bf16* Bp, int ldb) {
  const int lane  = threadIdx.x & 31;
  const int n     = lane & 15;
  const int kbase = (lane >> 4) << 4;         // 0 or 16
  v16bf b;
#pragma unroll
  for (int j = 0; j < 8; ++j) {
    b[2 * j]     = Bp[(size_t)(kbase + 2 * j) * ldb + n];
    b[2 * j + 1] = Bp[(size_t)(kbase + 2 * j + 1) * ldb + n];
  }
  return b;
}

// B fragment from an N x K row-major source (i.e. B = srcT). B[k][n]=src[n][k].
__device__ __forceinline__ v16bf load_bT_frag(const bf16* Bp, int ldb) {
  const int lane  = threadIdx.x & 31;
  const int n     = lane & 15;
  const int kbase = (lane >> 4) << 4;
  const bf16* row = Bp + (size_t)n * ldb + kbase;
  v16bf b;
#pragma unroll
  for (int j = 0; j < 8; ++j) {
    b[2 * j]     = row[2 * j];
    b[2 * j + 1] = row[2 * j + 1];
  }
  return b;
}

__device__ __forceinline__ v8f wmma_bf16(v16bf a, v16bf b, v8f c) {
  return __builtin_amdgcn_wmma_f32_16x16x32_bf16(
      /*neg_a=*/false, a, /*neg_b=*/false, b,
      /*c_mod=*/(short)0, c, /*reuse_a=*/false, /*reuse_b=*/false);
}

// --------------------------- block reductions ------------------------------
__device__ __forceinline__ float block_sum(float val) {
  __shared__ float tmp[9];
#pragma unroll
  for (int o = 16; o > 0; o >>= 1) val += __shfl_down(val, o, 32);
  __syncthreads();
  if ((threadIdx.x & 31) == 0) tmp[threadIdx.x >> 5] = val;
  __syncthreads();
  if (threadIdx.x == 0) {
    float s = 0.f;
#pragma unroll
    for (int i = 0; i < 8; ++i) s += tmp[i];
    tmp[8] = s;
  }
  __syncthreads();
  return tmp[8];
}

__device__ __forceinline__ float block_max(float val) {
  __shared__ float tmx[9];
#pragma unroll
  for (int o = 16; o > 0; o >>= 1) val = fmaxf(val, __shfl_down(val, o, 32));
  __syncthreads();
  if ((threadIdx.x & 31) == 0) tmx[threadIdx.x >> 5] = val;
  __syncthreads();
  if (threadIdx.x == 0) {
    float s = -3.4e38f;
#pragma unroll
    for (int i = 0; i < 8; ++i) s = fmaxf(s, tmx[i]);
    tmx[8] = s;
  }
  __syncthreads();
  return tmx[8];
}

// ----------------------------- small kernels -------------------------------
__global__ __launch_bounds__(256) void f32_to_bf16_kernel(
    const float* __restrict__ in, bf16* __restrict__ out, int n4) {
  const int i = blockIdx.x * 256 + threadIdx.x;
  if (i >= n4) return;
  const float4 v = ((const float4*)in)[i];
  bf16x4 o;
  o[0] = (bf16)v.x; o[1] = (bf16)v.y; o[2] = (bf16)v.z; o[3] = (bf16)v.w;
  *(bf16x4*)(out + (size_t)i * 4) = o;
}

// LayerNorm over D=512; one row per 256-thread block (2 elems/thread).
__global__ __launch_bounds__(256) void layernorm_kernel(
    const float* __restrict__ x, const float* __restrict__ g,
    const float* __restrict__ beta, bf16* __restrict__ out) {
  const size_t row = blockIdx.x;
  const float2 xv = ((const float2*)(x + row * cD))[threadIdx.x];
  const float mu = block_sum(xv.x + xv.y) * (1.0f / cD);
  const float d0 = xv.x - mu, d1 = xv.y - mu;
  const float var = block_sum(d0 * d0 + d1 * d1) * (1.0f / cD);
  const float rstd = rsqrtf(var + cEPS);
  const int c = threadIdx.x * 2;
  out[row * cD + c + 0] = (bf16)(d0 * rstd * g[c + 0] + beta[c + 0]);
  out[row * cD + c + 1] = (bf16)(d1 * rstd * g[c + 1] + beta[c + 1]);
}

// Per-row max and 1/sum(exp) over S=4096 score columns.
__global__ __launch_bounds__(256) void softmax_stats_kernel(
    const float* __restrict__ raw, float* __restrict__ rmax,
    float* __restrict__ rinv) {
  const size_t row = blockIdx.x;
  const float4* r = (const float4*)(raw + row * (size_t)cS);
  float4 v[4];
  float mx = -3.4e38f;
#pragma unroll
  for (int i = 0; i < 4; ++i) {
    v[i] = r[threadIdx.x + i * 256];
    mx = fmaxf(mx, fmaxf(fmaxf(v[i].x, v[i].y), fmaxf(v[i].z, v[i].w)));
  }
  mx = block_max(mx);
  float s = 0.f;
#pragma unroll
  for (int i = 0; i < 4; ++i)
    s += __expf(v[i].x - mx) + __expf(v[i].y - mx) +
         __expf(v[i].z - mx) + __expf(v[i].w - mx);
  s = block_sum(s);
  if (threadIdx.x == 0) { rmax[row] = mx; rinv[row] = 1.0f / s; }
}

// -------------------------- generic WMMA GEMM ------------------------------
// C[M,N] = epi(A[M,K] @ B[K,N] + bias[N] (+ resid)).
// Block = 256 threads = 8 waves (4 along M x 2 along N); block tile 128x64,
// each wave computes 32x32 (2x2 fragments -> 4 WMMAs per 32-deep k-step).
// Double-buffered LDS: tile i+1 DMA'd (TDM or async) while tile i computes.
// EPI: 0 = bf16 out; 1 = fp32 out + residual; 2 = bf16 out + exact-erf GELU.
template <int EPI>
__global__ __launch_bounds__(256) void gemm_bf16_kernel(
    const bf16* __restrict__ A, const bf16* __restrict__ Bm,
    const float* __restrict__ bias, const float* __restrict__ resid,
    void* __restrict__ outp, int M, int N, int K) {
  constexpr int LDA = 32 + 8, LDB = 64 + 8;
  __shared__ __align__(16) bf16 As[2][128][LDA];
  __shared__ __align__(16) bf16 Bs[2][32][LDB];
  const int tid = threadIdx.x, lane = tid & 31, wave = tid >> 5;
  const int wm = wave >> 1, wn = wave & 1;
  const int m0 = blockIdx.y * 128, n0 = blockIdx.x * 64;

  // fallback cooperative-load coordinates
  const int ar0 = tid >> 2, ac8 = (tid & 3) * 8;        // A: 2 tasks/thread
  const int br  = tid >> 3, bc8 = (tid & 7) * 8;        // B: 1 task/thread

  auto issue_tiles = [&](int k0, int buf) {
#if USE_TDM
    if (wave == 0) {        // A tile 128x32: rows of 64B -> interval 3, pad 3
      tdm_load_2d(A + (size_t)m0 * K + k0, lds_byte_addr(&As[buf][0][0]),
                  32, 128, (unsigned)K, 3, 3);
    } else if (wave == 1) { // B tile 32x64: rows of 128B -> interval 4, pad 3
      tdm_load_2d(Bm + (size_t)k0 * N + n0, lds_byte_addr(&Bs[buf][0][0]),
                  64, 32, (unsigned)N, 4, 3);
    }
#else
    copy16_g2s(A + (size_t)(m0 + ar0) * K + k0 + ac8, &As[buf][ar0][ac8]);
    copy16_g2s(A + (size_t)(m0 + ar0 + 64) * K + k0 + ac8,
               &As[buf][ar0 + 64][ac8]);
    copy16_g2s(Bm + (size_t)(k0 + br) * N + n0 + bc8, &Bs[buf][br][bc8]);
    if (k0 + 32 < K)
      __builtin_prefetch(A + (size_t)(m0 + ar0) * K + k0 + 32 + ac8, 0, 1);
#endif
  };

  issue_tiles(0, 0);
  int buf = 0;
  v8f acc[2][2] = {};
  for (int k0 = 0; k0 < K; k0 += 32) {
    wait_tiles();
    __syncthreads();
    if (k0 + 32 < K) issue_tiles(k0 + 32, buf ^ 1);
    const v16bf a0 = load_a_frag(&As[buf][wm * 32][0], LDA);
    const v16bf a1 = load_a_frag(&As[buf][wm * 32 + 16][0], LDA);
    const v16bf b0 = load_b_frag(&Bs[buf][0][wn * 32], LDB);
    const v16bf b1 = load_b_frag(&Bs[buf][0][wn * 32 + 16], LDB);
    acc[0][0] = wmma_bf16(a0, b0, acc[0][0]);
    acc[0][1] = wmma_bf16(a0, b1, acc[0][1]);
    acc[1][0] = wmma_bf16(a1, b0, acc[1][0]);
    acc[1][1] = wmma_bf16(a1, b1, acc[1][1]);
    buf ^= 1;
  }
  const int hi = lane >> 4, nn = lane & 15;
#pragma unroll
  for (int mi = 0; mi < 2; ++mi)
#pragma unroll
    for (int ni = 0; ni < 2; ++ni)
#pragma unroll
      for (int r = 0; r < 8; ++r) {
        const int rm = m0 + wm * 32 + mi * 16 + r + hi * 8;
        const int cn = n0 + wn * 32 + ni * 16 + nn;
        const size_t idx = (size_t)rm * N + cn;
        float v = acc[mi][ni][r] + bias[cn];
        if constexpr (EPI == 1) {
          ((float*)outp)[idx] = v + resid[idx];
        } else if constexpr (EPI == 2) {
          const float ge = 0.5f * v * (1.0f + erff(v * 0.70710678118654752f));
          ((bf16*)outp)[idx] = (bf16)ge;
        } else {
          ((bf16*)outp)[idx] = (bf16)v;
        }
      }
}

// ------------------------ attention: raw scores ----------------------------
// raw[b,h,qm,kn] = scale * (Qh @ KhT); block tile 128(q) x 64(k), K dim = 64.
// 8 waves = 4(M) x 2(N), 32x32 per wave, 4 WMMAs per wave per k-half.
__global__ __launch_bounds__(256) void attn_scores_kernel(
    const bf16* __restrict__ q, const bf16* __restrict__ k,
    float* __restrict__ raw) {
  constexpr int LDQ = cHD + 8;
  __shared__ __align__(16) bf16 Qs[128][LDQ];
  __shared__ __align__(16) bf16 Ks[64][LDQ];
  const int tid = threadIdx.x, lane = tid & 31, wave = tid >> 5;
  const int wm = wave >> 1, wn = wave & 1;
  const int bh = blockIdx.z, b = bh >> 3, h = bh & 7;
  const int m0 = blockIdx.y * 128;  // q rows
  const int n0 = blockIdx.x * 64;   // k rows
  const size_t head = ((size_t)b * cS) * cD + h * cHD;

  const int r8 = tid >> 3, c8 = (tid & 7) * 8;
#pragma unroll
  for (int i = 0; i < 4; ++i)       // Q tile 128x64
    copy16_g2s(q + head + (size_t)(m0 + r8 + i * 32) * cD + c8,
               &Qs[r8 + i * 32][c8]);
#pragma unroll
  for (int i = 0; i < 2; ++i)       // K tile 64x64
    copy16_g2s(k + head + (size_t)(n0 + r8 + i * 32) * cD + c8,
               &Ks[r8 + i * 32][c8]);
  wait_async_copies();
  __syncthreads();

  v8f acc[2][2] = {};
#pragma unroll
  for (int k0 = 0; k0 < cHD; k0 += 32) {
    const v16bf a0 = load_a_frag(&Qs[wm * 32][k0], LDQ);
    const v16bf a1 = load_a_frag(&Qs[wm * 32 + 16][k0], LDQ);
    const v16bf b0 = load_bT_frag(&Ks[wn * 32][k0], LDQ);
    const v16bf b1 = load_bT_frag(&Ks[wn * 32 + 16][k0], LDQ);
    acc[0][0] = wmma_bf16(a0, b0, acc[0][0]);
    acc[0][1] = wmma_bf16(a0, b1, acc[0][1]);
    acc[1][0] = wmma_bf16(a1, b0, acc[1][0]);
    acc[1][1] = wmma_bf16(a1, b1, acc[1][1]);
  }
  const float scale = 0.125f;       // HD^-0.5
  const int hi = lane >> 4, nn = lane & 15;
#pragma unroll
  for (int mi = 0; mi < 2; ++mi)
#pragma unroll
    for (int ni = 0; ni < 2; ++ni)
#pragma unroll
      for (int r = 0; r < 8; ++r) {
        const int qm = m0 + wm * 32 + mi * 16 + r + hi * 8;
        const int kn = n0 + wn * 32 + ni * 16 + nn;
        raw[((size_t)bh * cS + qm) * cS + kn] = acc[mi][ni][r] * scale;
      }
}

// ----------------- attention: normalize + ctx = P @ V ----------------------
// Each block owns 64 query rows of one (b,h): normalizes raw scores in place
// (final attn output, written exactly once) and accumulates ctx over S.
// 8 waves = 2(M) x 4(N); double-buffered P and V tiles, 2 WMMAs per k-step.
__global__ __launch_bounds__(256) void attn_ctx_kernel(
    float* __restrict__ attn, const bf16* __restrict__ vmat,
    const float* __restrict__ rmax, const float* __restrict__ rinv,
    bf16* __restrict__ ctx) {
  constexpr int LDP = 32 + 8, LDV = cHD + 8;
  __shared__ __align__(16) bf16 Ps[2][64][LDP];
  __shared__ __align__(16) bf16 Vs[2][32][LDV];
  const int tid = threadIdx.x, lane = tid & 31, wave = tid >> 5;
  const int wm = wave >> 2, wn = wave & 3;
  const int bh = blockIdx.y, b = bh >> 3, h = bh & 7;
  const int m0 = blockIdx.x * 64;

  const int pr = tid >> 3;                  // 0..31 (+32 for 2nd task)
  const int pc = (tid & 7) * 4;             // float4 col base
  const size_t prow0 = (size_t)bh * cS + m0;
  const float pm0 = rmax[prow0 + pr],      pi0 = rinv[prow0 + pr];
  const float pm1 = rmax[prow0 + pr + 32], pi1 = rinv[prow0 + pr + 32];
  const size_t head = ((size_t)b * cS) * cD + h * cHD;
  const int vc8 = (tid & 7) * 8;

  auto issue_v = [&](int kk0, int buf) {
#if USE_TDM
    if (wave == 0)          // V tile 32x64: rows of 128B -> interval 4, pad 3
      tdm_load_2d(vmat + head + (size_t)kk0 * cD, lds_byte_addr(&Vs[buf][0][0]),
                  64, 32, (unsigned)cD, 4, 3);
#else
    copy16_g2s(vmat + head + (size_t)(kk0 + pr) * cD + vc8, &Vs[buf][pr][vc8]);
#endif
  };

  // P tile 64x32: raw -> prob; write final attn; stash bf16 in LDS buffer.
  auto compute_p = [&](int kk0, int buf) {
#pragma unroll
    for (int i = 0; i < 2; ++i) {
      const int rr = pr + i * 32;
      const float pm = i ? pm1 : pm0, pinv = i ? pi1 : pi0;
      float* rowp = attn + (prow0 + rr) * (size_t)cS + kk0 + pc;
      const float4 rv = *(const float4*)rowp;
      float4 p;
      p.x = __expf(rv.x - pm) * pinv;
      p.y = __expf(rv.y - pm) * pinv;
      p.z = __expf(rv.z - pm) * pinv;
      p.w = __expf(rv.w - pm) * pinv;
      *(float4*)rowp = p;
      Ps[buf][rr][pc + 0] = (bf16)p.x;
      Ps[buf][rr][pc + 1] = (bf16)p.y;
      Ps[buf][rr][pc + 2] = (bf16)p.z;
      Ps[buf][rr][pc + 3] = (bf16)p.w;
    }
  };

  compute_p(0, 0);
  issue_v(0, 0);
  int buf = 0;
  v8f acc[2] = {};
  for (int kk0 = 0; kk0 < cS; kk0 += 32) {
    wait_tiles();
    __syncthreads();
    if (kk0 + 32 < cS) {
      issue_v(kk0 + 32, buf ^ 1);
      compute_p(kk0 + 32, buf ^ 1);
    }
    const v16bf a0 = load_a_frag(&Ps[buf][wm * 32][0], LDP);
    const v16bf a1 = load_a_frag(&Ps[buf][wm * 32 + 16][0], LDP);
    const v16bf bfr = load_b_frag(&Vs[buf][0][wn * 16], LDV);
    acc[0] = wmma_bf16(a0, bfr, acc[0]);
    acc[1] = wmma_bf16(a1, bfr, acc[1]);
    buf ^= 1;
  }
  const int hi = lane >> 4, nn = lane & 15;
#pragma unroll
  for (int mi = 0; mi < 2; ++mi)
#pragma unroll
    for (int r = 0; r < 8; ++r) {
      const int rm = m0 + wm * 32 + mi * 16 + r + hi * 8;
      const int cn = wn * 16 + nn;
      ctx[((size_t)b * cS + rm) * cD + h * cHD + cn] = (bf16)acc[mi][r];
    }
}

// ------------------------------- launcher ----------------------------------
extern "C" void kernel_launch(void* const* d_in, const int* in_sizes, int n_in,
                              void* d_out, int out_size, void* d_ws,
                              size_t ws_size, hipStream_t stream) {
  (void)in_sizes; (void)n_in; (void)out_size; (void)ws_size;
  const float* x    = (const float*)d_in[0];
  const float* Wq   = (const float*)d_in[1];
  const float* bq   = (const float*)d_in[2];
  const float* Wk   = (const float*)d_in[3];
  const float* bk   = (const float*)d_in[4];
  const float* Wv   = (const float*)d_in[5];
  const float* bv   = (const float*)d_in[6];
  const float* Wo   = (const float*)d_in[7];
  const float* bo   = (const float*)d_in[8];
  const float* W1   = (const float*)d_in[9];
  const float* b1   = (const float*)d_in[10];
  const float* W2   = (const float*)d_in[11];
  const float* b2   = (const float*)d_in[12];
  const float* g1   = (const float*)d_in[13];
  const float* be1  = (const float*)d_in[14];
  const float* g2   = (const float*)d_in[15];
  const float* be2  = (const float*)d_in[16];

  float* outX = (float*)d_out;                          // [B,S,D]
  float* attn = outX + (size_t)cB * cS * cD;            // [B,H,S,S]

  // workspace carve-out (256B aligned)
  char* base = (char*)d_ws;
  size_t off = 0;
  auto alloc = [&](size_t bytes) -> void* {
    void* r = base + off;
    off += (bytes + 255) & ~(size_t)255;
    return r;
  };
  bf16* wq_b  = (bf16*)alloc((size_t)cD * cD * 2);
  bf16* wk_b  = (bf16*)alloc((size_t)cD * cD * 2);
  bf16* wv_b  = (bf16*)alloc((size_t)cD * cD * 2);
  bf16* wo_b  = (bf16*)alloc((size_t)cD * cD * 2);
  bf16* w1_b  = (bf16*)alloc((size_t)cD * cFF * 2);
  bf16* w2_b  = (bf16*)alloc((size_t)cFF * cD * 2);
  bf16* h1    = (bf16*)alloc((size_t)ROWS * cD * 2);
  bf16* qb    = (bf16*)alloc((size_t)ROWS * cD * 2);
  bf16* kb    = (bf16*)alloc((size_t)ROWS * cD * 2);
  bf16* vb    = (bf16*)alloc((size_t)ROWS * cD * 2);
  bf16* ctxb  = (bf16*)alloc((size_t)ROWS * cD * 2);
  bf16* h2    = (bf16*)alloc((size_t)ROWS * cD * 2);
  bf16* ffn1  = (bf16*)alloc((size_t)ROWS * cFF * 2);
  float* x1   = (float*)alloc((size_t)ROWS * cD * 4);
  float* rmax = (float*)alloc((size_t)cB * cH * cS * 4);
  float* rinv = (float*)alloc((size_t)cB * cH * cS * 4);

  auto cvt = [&](const float* src, bf16* dst, int n) {
    const int n4 = n / 4;
    f32_to_bf16_kernel<<<(n4 + 255) / 256, 256, 0, stream>>>(src, dst, n4);
  };
  cvt(Wq, wq_b, cD * cD);
  cvt(Wk, wk_b, cD * cD);
  cvt(Wv, wv_b, cD * cD);
  cvt(Wo, wo_b, cD * cD);
  cvt(W1, w1_b, cD * cFF);
  cvt(W2, w2_b, cFF * cD);

  // --- attention block ---
  layernorm_kernel<<<ROWS, 256, 0, stream>>>(x, g1, be1, h1);

  const dim3 gDD(cD / 64, ROWS / 128);                  // (8,64)
  gemm_bf16_kernel<0><<<gDD, 256, 0, stream>>>(h1, wq_b, bq, nullptr, qb,
                                               ROWS, cD, cD);
  gemm_bf16_kernel<0><<<gDD, 256, 0, stream>>>(h1, wk_b, bk, nullptr, kb,
                                               ROWS, cD, cD);
  gemm_bf16_kernel<0><<<gDD, 256, 0, stream>>>(h1, wv_b, bv, nullptr, vb,
                                               ROWS, cD, cD);

  attn_scores_kernel<<<dim3(cS / 64, cS / 128, cB * cH), 256, 0, stream>>>(
      qb, kb, attn);
  softmax_stats_kernel<<<cB * cH * cS, 256, 0, stream>>>(attn, rmax, rinv);
  attn_ctx_kernel<<<dim3(cS / 64, cB * cH), 256, 0, stream>>>(
      attn, vb, rmax, rinv, ctxb);

  gemm_bf16_kernel<1><<<gDD, 256, 0, stream>>>(ctxb, wo_b, bo, x, x1,
                                               ROWS, cD, cD);

  // --- FFN block ---
  layernorm_kernel<<<ROWS, 256, 0, stream>>>(x1, g2, be2, h2);
  gemm_bf16_kernel<2><<<dim3(cFF / 64, ROWS / 128), 256, 0, stream>>>(
      h2, w1_b, b1, nullptr, ffn1, ROWS, cFF, cD);
  gemm_bf16_kernel<1><<<gDD, 256, 0, stream>>>(ffn1, w2_b, b2, x1, outX,
                                               ROWS, cD, cFF);
}
